// GINNet_46883863003469
// MI455X (gfx1250) — matrix operand order repaired and checked by reading, not compiled, not run on hardware
//
#include <hip/hip_runtime.h>

#define N_NODES  50000
#define N_EDGES  800000
#define HID      64
#define N_LAYERS 3

typedef __attribute__((ext_vector_type(16))) _Float16 v16h;
typedef __attribute__((ext_vector_type(8)))  _Float16 v8h;
typedef __attribute__((ext_vector_type(8)))  float    v8f;
typedef __attribute__((ext_vector_type(4)))  float    v4f;

static __device__ __forceinline__ v8f wmma_f16(v16h a, v16h b, v8f c) {
  // (neg_a, A, neg_b, B, c_mod, C, reuse_a, reuse_b)
  return __builtin_amdgcn_wmma_f32_16x16x32_f16(false, a, false, b, (short)0, c,
                                                false, false);
}

// Build a 16x32 f16 A-fragment (one K-block kb in {0,1}) from an f32 row.
// Per ISA 7.12.2: lanes 0-15 hold M=lane, K groups {kb*32+0..7, kb*32+16..23};
// lanes 16-31 hold the same rows with K offset +8 (hi = lane>>4).
static __device__ __forceinline__ v16h a_frag_f32row(const float* __restrict__ row,
                                                     int kb, int hi) {
  const float* p = row + kb * 32 + hi * 8;
  v4f a0 = *(const v4f*)(p + 0);
  v4f a1 = *(const v4f*)(p + 4);
  v4f a2 = *(const v4f*)(p + 16);
  v4f a3 = *(const v4f*)(p + 20);
  v16h f;
#pragma unroll
  for (int j = 0; j < 4; ++j) {
    f[j]      = (_Float16)a0[j];
    f[4 + j]  = (_Float16)a1[j];
    f[8 + j]  = (_Float16)a2[j];
    f[12 + j] = (_Float16)a3[j];
  }
  return f;
}

// Same A-fragment but straight from an f16 row: 2x b128 loads, zero converts.
static __device__ __forceinline__ v16h a_frag_f16row(const _Float16* __restrict__ row,
                                                     int kb, int hi) {
  const _Float16* p = row + kb * 32 + hi * 8;
  v8h r0 = *(const v8h*)(p);
  v8h r1 = *(const v8h*)(p + 16);
  v16h f;
#pragma unroll
  for (int j = 0; j < 8; ++j) { f[j] = r0[j]; f[8 + j] = r1[j]; }
  return f;
}

// Build a 32x16 f16 B-fragment from a k-major f32 weight block W[64][64].
// Assumed layout (matches A table order): lane holds column N=lane&15,
// elements j -> K = kb*32 + 8*(lane>=16) + j (+8 for j>=8).
static __device__ __forceinline__ v16h b_frag_f32(const float* __restrict__ W,
                                                  int kb, int nb, int lane) {
  const int n  = nb * 16 + (lane & 15);
  const int k0 = kb * 32 + 8 * (lane >> 4);
  v16h f;
#pragma unroll
  for (int j = 0; j < 16; ++j) f[j] = (_Float16)W[(k0 + j + (j >= 8 ? 8 : 0)) * HID + n];
  return f;
}

// ---------------------------------------------------------------------------
// h_out[N,64] = h_in[N,64] @ W[64,64] + b   (also writes the f16 mirror)
// ---------------------------------------------------------------------------
__global__ __launch_bounds__(256) void k_embed(const float* __restrict__ hin,
                                               const float* __restrict__ W,
                                               const float* __restrict__ bias,
                                               float* __restrict__ hout,
                                               _Float16* __restrict__ hout16,
                                               int ntiles) {
  const int lane = threadIdx.x & 31;
  const int hi = lane >> 4, ln = lane & 15;
  const int wave   = (blockIdx.x * blockDim.x + threadIdx.x) >> 5;
  const int nwaves = (gridDim.x * blockDim.x) >> 5;

  v16h Bf[2][4];
#pragma unroll
  for (int kb = 0; kb < 2; ++kb)
#pragma unroll
    for (int nb = 0; nb < 4; ++nb) Bf[kb][nb] = b_frag_f32(W, kb, nb, lane);
  float bv[4];
#pragma unroll
  for (int nb = 0; nb < 4; ++nb) bv[nb] = bias[nb * 16 + ln];

  const v8f vzero = {0.f, 0.f, 0.f, 0.f, 0.f, 0.f, 0.f, 0.f};
  for (int t = wave; t < ntiles; t += nwaves) {
    const float* row = hin + (size_t)(t * 16 + ln) * HID;
    v16h A0 = a_frag_f32row(row, 0, hi);
    v16h A1 = a_frag_f32row(row, 1, hi);
    v8f acc[4] = {vzero, vzero, vzero, vzero};
#pragma unroll
    for (int nb = 0; nb < 4; ++nb) {
      acc[nb] = wmma_f16(A0, Bf[0][nb], acc[nb]);
      acc[nb] = wmma_f16(A1, Bf[1][nb], acc[nb]);
    }
#pragma unroll
    for (int nb = 0; nb < 4; ++nb)
#pragma unroll
      for (int r = 0; r < 8; ++r) {
        const float v = acc[nb][r] + bv[nb];
        const size_t idx = (size_t)(t * 16 + r + 8 * hi) * HID + nb * 16 + ln;
        hout[idx]   = v;
        hout16[idx] = (_Float16)v;
      }
  }
}

// ---------------------------------------------------------------------------
// score[e] += relu([h[src]|h[dst]|e] @ W1 + b1) @ W2 + b2
// W1 is [130,64] k-major: rows 0-63 src part, 64-127 dst part, 128-129 edge.
// Gathers read the f16 mirror of h (half the L2 traffic, no converts).
// ---------------------------------------------------------------------------
__global__ __launch_bounds__(256) void k_pred(const _Float16* __restrict__ h16,
                                              const float* __restrict__ e,
                                              const int* __restrict__ src,
                                              const int* __restrict__ dst,
                                              const float* __restrict__ W1,
                                              const float* __restrict__ b1,
                                              const float* __restrict__ W2,
                                              const float* __restrict__ b2,
                                              float* __restrict__ score,
                                              int ntiles) {
  const int lane = threadIdx.x & 31;
  const int hi = lane >> 4, ln = lane & 15;
  const int wave   = (blockIdx.x * blockDim.x + threadIdx.x) >> 5;
  const int nwaves = (gridDim.x * blockDim.x) >> 5;

  v16h Bs[2][4], Bd[2][4];
#pragma unroll
  for (int kb = 0; kb < 2; ++kb)
#pragma unroll
    for (int nb = 0; nb < 4; ++nb) {
      Bs[kb][nb] = b_frag_f32(W1,            kb, nb, lane);
      Bd[kb][nb] = b_frag_f32(W1 + 64 * HID, kb, nb, lane);
    }
  float w1c0[4], w1c1[4], b1v[4], w2v[4];
#pragma unroll
  for (int nb = 0; nb < 4; ++nb) {
    const int n = nb * 16 + ln;
    w1c0[nb] = W1[128 * HID + n];
    w1c1[nb] = W1[129 * HID + n];
    b1v[nb]  = b1[n];
    w2v[nb]  = W2[n];
  }
  const float b2v = b2[0];

  const v8f vzero = {0.f, 0.f, 0.f, 0.f, 0.f, 0.f, 0.f, 0.f};
  for (int t = wave; t < ntiles; t += nwaves) {
    const int es = t * 16 + ln;
    // prefetch next tile's index cachelines (global_prefetch_b8)
    const int nt = t + nwaves;
    if (nt < ntiles) {
      __builtin_prefetch(src + nt * 16 + ln, 0, 1);
      __builtin_prefetch(dst + nt * 16 + ln, 0, 1);
    }
    const _Float16* rowS = h16 + (size_t)src[es] * HID;
    const _Float16* rowD = h16 + (size_t)dst[es] * HID;

    v8f acc[4] = {vzero, vzero, vzero, vzero};
    v16h A;
    A = a_frag_f16row(rowS, 0, hi);
#pragma unroll
    for (int nb = 0; nb < 4; ++nb) acc[nb] = wmma_f16(A, Bs[0][nb], acc[nb]);
    A = a_frag_f16row(rowS, 1, hi);
#pragma unroll
    for (int nb = 0; nb < 4; ++nb) acc[nb] = wmma_f16(A, Bs[1][nb], acc[nb]);
    A = a_frag_f16row(rowD, 0, hi);
#pragma unroll
    for (int nb = 0; nb < 4; ++nb) acc[nb] = wmma_f16(A, Bd[0][nb], acc[nb]);
    A = a_frag_f16row(rowD, 1, hi);
#pragma unroll
    for (int nb = 0; nb < 4; ++nb) acc[nb] = wmma_f16(A, Bd[1][nb], acc[nb]);

    // per-lane rows M = r + 8*hi -> edges t*16 + 8*hi + r ; e is [E,2]
    const float* ep = e + (size_t)(t * 16 + 8 * hi) * 2;
    v4f e01 = *(const v4f*)(ep + 0);
    v4f e23 = *(const v4f*)(ep + 4);
    v4f e45 = *(const v4f*)(ep + 8);
    v4f e67 = *(const v4f*)(ep + 12);
    float e0[8] = {e01[0], e01[2], e23[0], e23[2], e45[0], e45[2], e67[0], e67[2]};
    float e1[8] = {e01[1], e01[3], e23[1], e23[3], e45[1], e45[3], e67[1], e67[3]};

    float val[8] = {0.f, 0.f, 0.f, 0.f, 0.f, 0.f, 0.f, 0.f};
#pragma unroll
    for (int nb = 0; nb < 4; ++nb)
#pragma unroll
      for (int r = 0; r < 8; ++r) {
        float z = acc[nb][r] + b1v[nb] + e0[r] * w1c0[nb] + e1[r] * w1c1[nb];
        z = fmaxf(z, 0.f);
        val[r] += z * w2v[nb];
      }
    // reduce over the 16 lanes sharing each output row (half-wave groups)
#pragma unroll
    for (int r = 0; r < 8; ++r) {
      val[r] += __shfl_xor(val[r], 1, 32);
      val[r] += __shfl_xor(val[r], 2, 32);
      val[r] += __shfl_xor(val[r], 4, 32);
      val[r] += __shfl_xor(val[r], 8, 32);
    }
    if (ln == 0) {
      float* sp = score + t * 16 + 8 * hi;
#pragma unroll
      for (int r = 0; r < 8; ++r) sp[r] += val[r] + b2v;
    }
  }
}

// ---------------------------------------------------------------------------
// neigh[dst] += h[src]  (GIN sum aggregation); thread = (edge, 4-feature chunk)
// ---------------------------------------------------------------------------
__global__ void k_aggregate(const float* __restrict__ h,
                            const int* __restrict__ src,
                            const int* __restrict__ dst,
                            float* __restrict__ neigh) {
  const long tid = (long)blockIdx.x * blockDim.x + threadIdx.x;
  if (tid >= (long)N_EDGES * 16) return;
  const int edge = (int)(tid >> 4);
  const int c4   = (int)(tid & 15) << 2;
  v4f v = *(const v4f*)(h + (size_t)src[edge] * HID + c4);
  float* out = neigh + (size_t)dst[edge] * HID + c4;
  atomicAdd(out + 0, v[0]);
  atomicAdd(out + 1, v[1]);
  atomicAdd(out + 2, v[2]);
  atomicAdd(out + 3, v[3]);
}

// ---------------------------------------------------------------------------
// x = (1+eps)*h + neigh; z1 = relu(bn1(x@W1+b1)); z2 = z1@W2+b2;
// x = relu(bn_gin(relu(bn_apply(z2)))); h += x  (and refresh the f16 mirror)
// ---------------------------------------------------------------------------
__global__ __launch_bounds__(256) void k_node_mlp(
    float* __restrict__ h, _Float16* __restrict__ h16,
    const float* __restrict__ neigh,
    const float* __restrict__ eps,
    const float* __restrict__ W1, const float* __restrict__ b1,
    const float* __restrict__ bn1s, const float* __restrict__ bn1sh,
    const float* __restrict__ W2, const float* __restrict__ b2,
    const float* __restrict__ abns, const float* __restrict__ abnsh,
    const float* __restrict__ gbns, const float* __restrict__ gbnsh,
    int ntiles) {
  __shared__ __align__(16) _Float16 zbuf[8][16 * HID];  // per-wave 16x64 f16 stage
  const int lane = threadIdx.x & 31;
  const int hi = lane >> 4, ln = lane & 15;
  const int wslot  = threadIdx.x >> 5;
  const int wave   = (blockIdx.x * blockDim.x + threadIdx.x) >> 5;
  const int nwaves = (gridDim.x * blockDim.x) >> 5;
  const float epsv = 1.0f + *eps;

  v16h B1[2][4], B2[2][4];
#pragma unroll
  for (int kb = 0; kb < 2; ++kb)
#pragma unroll
    for (int nb = 0; nb < 4; ++nb) {
      B1[kb][nb] = b_frag_f32(W1, kb, nb, lane);
      B2[kb][nb] = b_frag_f32(W2, kb, nb, lane);
    }
  float b1v[4], s1[4], sh1[4], b2v[4], as_[4], ash[4], gs[4], gsh[4];
#pragma unroll
  for (int nb = 0; nb < 4; ++nb) {
    const int n = nb * 16 + ln;
    b1v[nb] = b1[n];  s1[nb] = bn1s[n];  sh1[nb] = bn1sh[n];
    b2v[nb] = b2[n];  as_[nb] = abns[n]; ash[nb] = abnsh[n];
    gs[nb] = gbns[n]; gsh[nb] = gbnsh[n];
  }
  _Float16* zb = zbuf[wslot];
  const v8f vzero = {0.f, 0.f, 0.f, 0.f, 0.f, 0.f, 0.f, 0.f};

  for (int t = wave; t < ntiles; t += nwaves) {
    const float* rowH = h + (size_t)(t * 16 + ln) * HID;
    const float* rowN = neigh + (size_t)(t * 16 + ln) * HID;

    v8f acc[4] = {vzero, vzero, vzero, vzero};
#pragma unroll
    for (int kb = 0; kb < 2; ++kb) {
      const float* ph = rowH + kb * 32 + hi * 8;
      const float* pn = rowN + kb * 32 + hi * 8;
      v4f h0 = *(const v4f*)(ph + 0),  h1 = *(const v4f*)(ph + 4);
      v4f h2 = *(const v4f*)(ph + 16), h3 = *(const v4f*)(ph + 20);
      v4f n0 = *(const v4f*)(pn + 0),  n1 = *(const v4f*)(pn + 4);
      v4f n2 = *(const v4f*)(pn + 16), n3 = *(const v4f*)(pn + 20);
      v16h A;
#pragma unroll
      for (int j = 0; j < 4; ++j) {
        A[j]      = (_Float16)(epsv * h0[j] + n0[j]);
        A[4 + j]  = (_Float16)(epsv * h1[j] + n1[j]);
        A[8 + j]  = (_Float16)(epsv * h2[j] + n2[j]);
        A[12 + j] = (_Float16)(epsv * h3[j] + n3[j]);
      }
#pragma unroll
      for (int nb = 0; nb < 4; ++nb) acc[nb] = wmma_f16(A, B1[kb][nb], acc[nb]);
    }
    // z1 = relu(bn1(. + b1)) -> LDS in [m][k] f16 layout for the transposed reload
#pragma unroll
    for (int nb = 0; nb < 4; ++nb)
#pragma unroll
      for (int r = 0; r < 8; ++r) {
        float z = (acc[nb][r] + b1v[nb]) * s1[nb] + sh1[nb];
        zb[(r + 8 * hi) * HID + nb * 16 + ln] = (_Float16)fmaxf(z, 0.f);
      }
    asm volatile("s_wait_dscnt 0" ::: "memory");  // cross-lane LDS RAW

    v8f acc2[4] = {vzero, vzero, vzero, vzero};
#pragma unroll
    for (int kb = 0; kb < 2; ++kb) {
      const _Float16* zr = zb + ln * HID + kb * 32 + hi * 8;
      v8h r0 = *(const v8h*)(zr);
      v8h r1 = *(const v8h*)(zr + 16);
      v16h A2;
#pragma unroll
      for (int j = 0; j < 8; ++j) { A2[j] = r0[j]; A2[8 + j] = r1[j]; }
#pragma unroll
      for (int nb = 0; nb < 4; ++nb) acc2[nb] = wmma_f16(A2, B2[kb][nb], acc2[nb]);
    }
    // epilogue: +b2, apply-BN+relu, gin-BN+relu, residual add into h (+mirror)
#pragma unroll
    for (int nb = 0; nb < 4; ++nb)
#pragma unroll
      for (int r = 0; r < 8; ++r) {
        float z = acc2[nb][r] + b2v[nb];
        z = fmaxf(z * as_[nb] + ash[nb], 0.f);
        z = fmaxf(z * gs[nb] + gsh[nb], 0.f);
        const size_t idx = (size_t)(t * 16 + r + 8 * hi) * HID + nb * 16 + ln;
        const float nv = h[idx] + z;
        h[idx]   = nv;
        h16[idx] = (_Float16)nv;
      }
  }
}

__global__ void k_relu(float* __restrict__ s, int n) {
  const int i = blockIdx.x * blockDim.x + threadIdx.x;
  if (i < n) s[i] = fmaxf(s[i], 0.f);
}

// ---------------------------------------------------------------------------
extern "C" void kernel_launch(void* const* d_in, const int* in_sizes, int n_in,
                              void* d_out, int out_size, void* d_ws, size_t ws_size,
                              hipStream_t stream) {
  (void)in_sizes; (void)n_in; (void)out_size; (void)ws_size;
  const float* h_in   = (const float*)d_in[0];
  const float* e      = (const float*)d_in[1];
  const int*   src    = (const int*)d_in[2];
  const int*   dst    = (const int*)d_in[3];
  const float* emb_W  = (const float*)d_in[4];
  const float* emb_b  = (const float*)d_in[5];
  const float* eps    = (const float*)d_in[6];
  const float* mW1    = (const float*)d_in[7];
  const float* mb1    = (const float*)d_in[8];
  const float* bn1s   = (const float*)d_in[9];
  const float* bn1sh  = (const float*)d_in[10];
  const float* mW2    = (const float*)d_in[11];
  const float* mb2    = (const float*)d_in[12];
  const float* abns   = (const float*)d_in[13];
  const float* abnsh  = (const float*)d_in[14];
  const float* gbns   = (const float*)d_in[15];
  const float* gbnsh  = (const float*)d_in[16];
  const float* pW1    = (const float*)d_in[17];
  const float* pb1    = (const float*)d_in[18];
  const float* pW2    = (const float*)d_in[19];
  const float* pb2    = (const float*)d_in[20];
  float* score = (float*)d_out;

  float*    h     = (float*)d_ws;                  // [N,64] f32 master
  float*    neigh = h + (size_t)N_NODES * HID;     // [N,64] f32 scratch
  _Float16* h16   = (_Float16*)(neigh + (size_t)N_NODES * HID);  // f16 mirror

  const int node_tiles = N_NODES / 16;   // 3125 (exact)
  const int edge_tiles = N_EDGES / 16;   // 50000 (exact)

  hipMemsetAsync(score, 0, (size_t)N_EDGES * sizeof(float), stream);

  k_embed<<<128, 256, 0, stream>>>(h_in, emb_W, emb_b, h, h16, node_tiles);
  k_pred<<<512, 256, 0, stream>>>(h16, e, src, dst, pW1, pb1, pW2, pb2, score,
                                  edge_tiles);
  for (int i = 0; i < N_LAYERS; ++i) {
    hipMemsetAsync(neigh, 0, (size_t)N_NODES * HID * sizeof(float), stream);
    k_aggregate<<<(N_EDGES * 16) / 256, 256, 0, stream>>>(h, src, dst, neigh);
    k_node_mlp<<<128, 256, 0, stream>>>(h, h16, neigh, eps + i,
                                        mW1 + (size_t)i * HID * HID, mb1 + i * HID,
                                        bn1s + i * HID, bn1sh + i * HID,
                                        mW2 + (size_t)i * HID * HID, mb2 + i * HID,
                                        abns + i * HID, abnsh + i * HID,
                                        gbns + i * HID, gbnsh + i * HID, node_tiles);
    k_pred<<<512, 256, 0, stream>>>(h16, e, src, dst,
                                    pW1 + (size_t)(i + 1) * 130 * HID,
                                    pb1 + (i + 1) * HID,
                                    pW2 + (i + 1) * HID, pb2 + (i + 1),
                                    score, edge_tiles);
  }
  k_relu<<<(N_EDGES + 255) / 256, 256, 0, stream>>>(score, N_EDGES);
}
